// KGraphSAINT_36155034697969
// MI455X (gfx1250) — compile-verified, matrix operand order checked by reading, not run on hardware
//
#include <hip/hip_runtime.h>
#include <hip/hip_bf16.h>

typedef __bf16 bf16_t;
typedef __attribute__((ext_vector_type(16))) __bf16 v16bf;
typedef __attribute__((ext_vector_type(8)))  float  v8f;

#define DIM   64
#define NNB   16
#define TB    16                 // batch elements per workgroup
#define NREL  33                 // NUM_REL + 1
#define ROWS1 (TB + TB*NNB)      // 272 rows for iteration-0 matmul
#define NTHREADS 256

__device__ __forceinline__ bf16_t f2bf(float f) {
  union { float f; unsigned u; } in; in.f = f;
  unsigned x = in.u;
  unsigned r = (x + 0x7FFFu + ((x >> 16) & 1u)) >> 16;   // RNE
  union { unsigned short s; bf16_t b; } out; out.s = (unsigned short)r;
  return out.b;
}
__device__ __forceinline__ float bf2f(bf16_t b) {
  union { unsigned short s; bf16_t b; } in; in.b = b;
  union { unsigned u; float f; } out; out.u = ((unsigned)in.s) << 16;
  return out.f;
}
// fast hardware reciprocal (v_rcp_f32) instead of IEEE divide expansion
__device__ __forceinline__ float frcp_(float x)  { return __builtin_amdgcn_rcpf(x); }
__device__ __forceinline__ float sigmoidf_(float x) { return frcp_(1.0f + __expf(-x)); }
__device__ __forceinline__ float tanhf_(float x) {
  float e2 = __expf(2.0f * x);                 // v_exp_f32
  return (e2 - 1.0f) * frcp_(e2 + 1.0f);       // v_rcp_f32
}

__global__ __launch_bounds__(NTHREADS)
void kgnn_kernel(const int* __restrict__ u, const int* __restrict__ v,
                 const int* __restrict__ adj, const int* __restrict__ rel_adj,
                 const float* __restrict__ usr_emb, const float* __restrict__ ent_emb,
                 const float* __restrict__ rel_emb,
                 const float* __restrict__ W0, const float* __restrict__ b0,
                 const float* __restrict__ W1, const float* __restrict__ b1,
                 float* __restrict__ out)
{
  __shared__ float  s_user[TB*DIM];          // user embeddings f32
  __shared__ float  s_relc[NREL*DIM];        // rel_emb cache f32
  __shared__ float  s_S[TB*NREL];            // user . rel_emb[r] score table
  __shared__ int    s_e0[TB];                // hop-0 entities
  __shared__ int    s_e1[TB*NNB];            // hop-1 entities
  __shared__ int    s_r1[TB*NNB];            // rels e0->e1
  __shared__ float  s_attn0[TB*NNB];         // softmax weights hop0 (reused iter1)
  __shared__ float  s_attn1[TB*NNB*NNB];     // softmax weights hop1
  __shared__ int    s_e2[TB*NNB*NNB];        // hop-2 entities
  __shared__ bf16_t s_X[ROWS1*DIM];          // matmul A input (bf16)
  __shared__ bf16_t s_Y[ROWS1*DIM];          // matmul output after sigmoid (bf16)
  __shared__ bf16_t s_W[DIM*DIM];            // weights bf16 (W0 then W1)
  __shared__ float  s_bias[DIM];
  __shared__ float  s_item[TB*DIM];          // final item vectors f32

  const int t    = threadIdx.x;
  const int gb0  = blockIdx.x * TB;
  const int lane = t & 31;
  const int wave = t >> 5;

  // ---------- Stage A: cooperative loads ----------
  if (t < TB) s_e0[t] = v[gb0 + t];
  {
    int row = t >> 4, dg = t & 15;
    int uid = u[gb0 + row];
    reinterpret_cast<float4*>(s_user)[row*16 + dg] =
        reinterpret_cast<const float4*>(usr_emb + uid*DIM)[dg];
  }
  for (int idx = t; idx < NREL*DIM/4; idx += NTHREADS)
    reinterpret_cast<float4*>(s_relc)[idx] = reinterpret_cast<const float4*>(rel_emb)[idx];
  #pragma unroll
  for (int k = 0; k < 16; ++k) s_W[t*16 + k] = f2bf(W0[t*16 + k]);
  if (t < DIM) s_bias[t] = b0[t];
  __syncthreads();

  // ---------- Stage B: hop-1 indices ----------
  {
    int e = s_e0[t >> 4];
    s_e1[t] = adj[e*NNB + (t & 15)];
    s_r1[t] = rel_adj[e*NNB + (t & 15)];
  }
  // ---------- Stage C: user-relation score table ----------
  for (int idx = t; idx < TB*NREL; idx += NTHREADS) {
    int b = idx / NREL, r = idx % NREL;
    const float* ub = s_user + b*DIM;
    const float* rb = s_relc + r*DIM;
    float acc = 0.f;
    #pragma unroll
    for (int d = 0; d < DIM; ++d) acc += ub[d]*rb[d];
    s_S[idx] = acc;
  }
  __syncthreads();

  // ---------- attention weights: hop0 (one thread per b) ----------
  if (t < TB) {
    int b = t;
    float s[NNB], m = -1e30f;
    #pragma unroll
    for (int n = 0; n < NNB; ++n) { s[n] = s_S[b*NREL + s_r1[b*NNB+n]]; m = fmaxf(m, s[n]); }
    float sum = 0.f;
    #pragma unroll
    for (int n = 0; n < NNB; ++n) { s[n] = __expf(s[n]-m); sum += s[n]; }
    float inv = frcp_(sum);
    #pragma unroll
    for (int n = 0; n < NNB; ++n) s_attn0[b*NNB+n] = s[n]*inv;
  }
  // ---------- attention weights + neighbor ids: hop1 (one thread per (b,m)) ----------
  {
    int p = t, b = p >> 4;
    int e1p = s_e1[p];
    float s[NNB], m = -1e30f;
    #pragma unroll
    for (int n = 0; n < NNB; ++n) {
      int e2 = adj[e1p*NNB + n];
      int r2 = rel_adj[e1p*NNB + n];
      s_e2[p*NNB+n] = e2;
      s[n] = s_S[b*NREL + r2];
      m = fmaxf(m, s[n]);
    }
    float sum = 0.f;
    #pragma unroll
    for (int n = 0; n < NNB; ++n) { s[n] = __expf(s[n]-m); sum += s[n]; }
    float inv = frcp_(sum);
    #pragma unroll
    for (int n = 0; n < NNB; ++n) s_attn1[p*NNB+n] = s[n]*inv;
  }
  __syncthreads();

  // ---------- build X rows 0..15 : sv0 + agg(e1) ; coalesced over dims ----------
  {
    int b = t >> 4, dg = t & 15;
    float4 acc = reinterpret_cast<const float4*>(ent_emb + s_e0[b]*DIM)[dg];
    #pragma unroll
    for (int n = 0; n < NNB; ++n) {
      float a = s_attn0[b*NNB+n];
      float4 nv = reinterpret_cast<const float4*>(ent_emb + s_e1[b*NNB+n]*DIM)[dg];
      acc.x += a*nv.x; acc.y += a*nv.y; acc.z += a*nv.z; acc.w += a*nv.w;
    }
    int base = b*DIM + dg*4;
    s_X[base+0]=f2bf(acc.x); s_X[base+1]=f2bf(acc.y); s_X[base+2]=f2bf(acc.z); s_X[base+3]=f2bf(acc.w);
  }
  // ---------- build X rows 16..271 : sv1 + agg(e2) ----------
  {
    int g = t >> 4, dg = t & 15;
    for (int k = 0; k < 16; ++k) {
      int p = k*16 + g;                       // 16 consecutive lanes share p -> coalesced
      float4 acc = reinterpret_cast<const float4*>(ent_emb + s_e1[p]*DIM)[dg];
      #pragma unroll
      for (int n = 0; n < NNB; ++n) {
        float a = s_attn1[p*NNB+n];
        float4 nv = reinterpret_cast<const float4*>(ent_emb + s_e2[p*NNB+n]*DIM)[dg];
        acc.x += a*nv.x; acc.y += a*nv.y; acc.z += a*nv.z; acc.w += a*nv.w;
      }
      int base = (TB + p)*DIM + dg*4;
      s_X[base+0]=f2bf(acc.x); s_X[base+1]=f2bf(acc.y); s_X[base+2]=f2bf(acc.z); s_X[base+3]=f2bf(acc.w);
    }
  }
  __syncthreads();

  // ---------- WMMA stage 1: Y = sigmoid(X[272x64] @ W0 + b0) ----------
  {
    const int rr = lane & 15;
    const int h  = lane >> 4;
    for (int tile = wave; tile < (ROWS1/16)*4; tile += 8) {
      int mt = tile >> 2, nt = tile & 3;
      v16bf a0, a1, bv0, bv1;
      #pragma unroll
      for (int j = 0; j < 16; ++j) {
        int kk = (j < 8) ? (h*8 + j) : (16 + h*8 + (j - 8));   // ISA 16-bit A/B K map
        a0[j]  = s_X[(mt*16 + rr)*DIM + kk];
        a1[j]  = s_X[(mt*16 + rr)*DIM + 32 + kk];
        bv0[j] = s_W[kk*DIM + nt*16 + rr];
        bv1[j] = s_W[(32+kk)*DIM + nt*16 + rr];
      }
      v8f c = {};
      c = __builtin_amdgcn_wmma_f32_16x16x32_bf16(false, a0, false, bv0, (short)0, c, false, false);
      c = __builtin_amdgcn_wmma_f32_16x16x32_bf16(false, a1, false, bv1, (short)0, c, false, false);
      int col = nt*16 + rr;
      float bias = s_bias[col];
      #pragma unroll
      for (int i = 0; i < 8; ++i) {
        int row = mt*16 + i + 8*h;
        s_Y[row*DIM + col] = f2bf(sigmoidf_(c[i] + bias));
      }
    }
  }
  __syncthreads();

  // ---------- iteration 1 hop 0: reuse attn0; X row b = Y[b] + sum attn0*Y[16+b*16+n] ----------
  #pragma unroll
  for (int k = 0; k < 16; ++k) s_W[t*16 + k] = f2bf(W1[t*16 + k]);
  if (t < DIM) s_bias[t] = b1[t];
  {
    int b = t >> 4, dg = t & 15;
    #pragma unroll
    for (int q = 0; q < 4; ++q) {
      int d = dg*4 + q;
      float acc = bf2f(s_Y[b*DIM + d]);
      #pragma unroll
      for (int n = 0; n < NNB; ++n)
        acc += s_attn0[b*NNB+n] * bf2f(s_Y[(TB + b*NNB + n)*DIM + d]);
      s_X[b*DIM + d] = f2bf(acc);
    }
  }
  __syncthreads();

  // ---------- WMMA stage 2: item = tanh(X[16x64] @ W1 + b1) ----------
  if (wave < 4) {
    const int rr = lane & 15;
    const int h  = lane >> 4;
    int nt = wave;
    v16bf a0, a1, bv0, bv1;
    #pragma unroll
    for (int j = 0; j < 16; ++j) {
      int kk = (j < 8) ? (h*8 + j) : (16 + h*8 + (j - 8));
      a0[j]  = s_X[rr*DIM + kk];
      a1[j]  = s_X[rr*DIM + 32 + kk];
      bv0[j] = s_W[kk*DIM + nt*16 + rr];
      bv1[j] = s_W[(32+kk)*DIM + nt*16 + rr];
    }
    v8f c = {};
    c = __builtin_amdgcn_wmma_f32_16x16x32_bf16(false, a0, false, bv0, (short)0, c, false, false);
    c = __builtin_amdgcn_wmma_f32_16x16x32_bf16(false, a1, false, bv1, (short)0, c, false, false);
    int col = nt*16 + rr;
    float bias = s_bias[col];
    #pragma unroll
    for (int i = 0; i < 8; ++i) {
      int row = i + 8*h;
      s_item[row*DIM + col] = tanhf_(c[i] + bias);
    }
  }
  __syncthreads();

  // ---------- final: sigmoid(user . item) ----------
  if (t < TB) {
    const float* ub = s_user + t*DIM;
    const float* ib = s_item + t*DIM;
    float acc = 0.f;
    #pragma unroll
    for (int d = 0; d < DIM; ++d) acc += ub[d]*ib[d];
    out[gb0 + t] = sigmoidf_(acc);
  }
}

extern "C" void kernel_launch(void* const* d_in, const int* in_sizes, int n_in,
                              void* d_out, int out_size, void* d_ws, size_t ws_size,
                              hipStream_t stream) {
  const int*   u       = (const int*)d_in[0];
  const int*   v       = (const int*)d_in[1];
  const int*   adj     = (const int*)d_in[2];
  const int*   rel_adj = (const int*)d_in[3];
  const float* usr_emb = (const float*)d_in[4];
  const float* ent_emb = (const float*)d_in[5];
  const float* rel_emb = (const float*)d_in[6];
  const float* W0      = (const float*)d_in[7];
  const float* b0      = (const float*)d_in[8];
  const float* W1      = (const float*)d_in[9];
  const float* b1      = (const float*)d_in[10];
  float* out = (float*)d_out;

  int B = in_sizes[0];           // BATCH = 4096
  int nblocks = B / TB;          // 256 workgroups
  kgnn_kernel<<<nblocks, NTHREADS, 0, stream>>>(u, v, adj, rel_adj,
                                                usr_emb, ent_emb, rel_emb,
                                                W0, b0, W1, b1, out);
}
